// MultiHeadAttention_13211319403251
// MI455X (gfx1250) — matrix operand order repaired and checked by reading, not compiled, not run on hardware
//
#include <hip/hip_runtime.h>
#include <stdint.h>

// ---------------- problem constants ----------------
#define BDIM   2
#define LLEN   2048
#define DDIM   1024
#define HHEADS 16
#define DKDIM  64
#define DVDIM  64
#define TOPKN  32

typedef __attribute__((ext_vector_type(16))) __bf16 v16bf;
typedef __attribute__((ext_vector_type(8)))  float  v8f;

// ---------------- helpers ----------------
__device__ __forceinline__ unsigned short f32_to_bf16(float f) {
  unsigned int u = __float_as_uint(f);
  unsigned int r = (u + 0x7fffu + ((u >> 16) & 1u)) >> 16;   // RNE
  return (unsigned short)r;
}
__device__ __forceinline__ __bf16 us_bf(unsigned short u) {
  return __builtin_bit_cast(__bf16, u);
}

__device__ __forceinline__ v8f wmma_bf16(v16bf a, v16bf b, v8f c) {
  // v_wmma_f32_16x16x32_bf16: (neg_a, A, neg_b, B, c_mod, C, reuse_a, reuse_b)
  return __builtin_amdgcn_wmma_f32_16x16x32_bf16(false, a, false, b, (short)0, c,
                                                 false, false);
}

// A fragment 16x32 bf16 from LDS tile, row-major with leading dim ldk.
// lane<16: M=lane, K groups {0..7,16..23}; lane>=16: M=lane-16, K {8..15,24..31}
__device__ __forceinline__ v16bf frag_a(const unsigned short* lds, int m0, int ldk,
                                        int lane) {
  const unsigned short* p = lds + (m0 + (lane & 15)) * ldk;
  const int kb = (lane >> 4) << 3;  // 0 or 8
  v16bf a;
#pragma unroll
  for (int j = 0; j < 8; ++j) a[j] = us_bf(p[kb + j]);
#pragma unroll
  for (int j = 0; j < 8; ++j) a[8 + j] = us_bf(p[16 + kb + j]);
  return a;
}

// B fragment 32x16 bf16 from LDS tile [K][N] row-major, leading dim ldn.
// lane<16: N=lane, K groups {0..7,16..23}; lane>=16: N=lane-16, K {8..15,24..31}
__device__ __forceinline__ v16bf frag_b(const unsigned short* lds, int n0, int ldn,
                                        int lane) {
  const unsigned short* p = lds + n0 + (lane & 15);
  const int kb = (lane >> 4) << 3;  // 0 or 8
  v16bf b;
#pragma unroll
  for (int j = 0; j < 8; ++j) b[j] = us_bf(p[(kb + j) * ldn]);
#pragma unroll
  for (int j = 0; j < 8; ++j) b[8 + j] = us_bf(p[(16 + kb + j) * ldn]);
  return b;
}

// ---------------- generic 128x128 GEMM, bf16 WMMA, f32 accumulate -------------
// A: [M x K] row-major (f32 or bf16), W: [K x N] row-major f32,
// C: [M x N] (bf16 or f32). 256 threads = 8 waves; wave tile 64x32 (4x2 wmma).
template <bool A_BF16, bool OUT_F32>
__global__ __launch_bounds__(256) void gemm128x128(const void* __restrict__ Av,
                                                   const float* __restrict__ W,
                                                   void* __restrict__ Cv, int M, int N,
                                                   int K, int lda, int ldb, int ldc) {
  __shared__ unsigned short sA[128 * 32];
  __shared__ unsigned short sB[32 * 128];
  const int tid  = threadIdx.x;
  const int lane = tid & 31;
  const int w    = tid >> 5;
  const int wm   = (w & 1) * 64;
  const int wn   = (w >> 1) * 32;
  const int m0   = blockIdx.y * 128;
  const int n0   = blockIdx.x * 128;

  v8f acc[4][2];
#pragma unroll
  for (int mi = 0; mi < 4; ++mi)
#pragma unroll
    for (int ni = 0; ni < 2; ++ni) acc[mi][ni] = (v8f){0, 0, 0, 0, 0, 0, 0, 0};

  for (int k0 = 0; k0 < K; k0 += 32) {
    __syncthreads();
#pragma unroll
    for (int i = tid; i < 128 * 32; i += 256) {
      const int r = i >> 5, c = i & 31;
      if constexpr (A_BF16)
        sA[i] = ((const unsigned short*)Av)[(size_t)(m0 + r) * lda + k0 + c];
      else
        sA[i] = f32_to_bf16(((const float*)Av)[(size_t)(m0 + r) * lda + k0 + c]);
    }
#pragma unroll
    for (int i = tid; i < 32 * 128; i += 256) {
      const int r = i >> 7, c = i & 127;
      sB[i] = f32_to_bf16(W[(size_t)(k0 + r) * ldb + n0 + c]);
    }
    __syncthreads();

    v16bf af[4], bfr[2];
#pragma unroll
    for (int mi = 0; mi < 4; ++mi) af[mi] = frag_a(sA, wm + mi * 16, 32, lane);
#pragma unroll
    for (int ni = 0; ni < 2; ++ni) bfr[ni] = frag_b(sB, wn + ni * 16, 128, lane);
#pragma unroll
    for (int mi = 0; mi < 4; ++mi)
#pragma unroll
      for (int ni = 0; ni < 2; ++ni) acc[mi][ni] = wmma_bf16(af[mi], bfr[ni], acc[mi][ni]);
  }

  const int cn    = n0 + wn + (lane & 15);
  const int rbase = m0 + wm + ((lane >> 4) << 3);
#pragma unroll
  for (int mi = 0; mi < 4; ++mi)
#pragma unroll
    for (int ni = 0; ni < 2; ++ni)
#pragma unroll
      for (int r = 0; r < 8; ++r) {
        const size_t row = rbase + mi * 16 + r;
        const size_t col = cn + ni * 16;
        if constexpr (OUT_F32)
          ((float*)Cv)[row * ldc + col] = acc[mi][ni][r];
        else
          ((unsigned short*)Cv)[row * ldc + col] = f32_to_bf16(acc[mi][ni][r]);
      }
}

// ---------------- scores: S = (Qh Kh^T)/8 per (b,h), 128x128 tiles ------------
__global__ __launch_bounds__(256) void attn_scores_kernel(
    const unsigned short* __restrict__ qh, const unsigned short* __restrict__ kh,
    float* __restrict__ scores, float scale) {
  __shared__ unsigned short sA[128 * 64];   // Qh rows  [m][d]
  __shared__ unsigned short sB[64 * 128];   // Kh^T     [d][n]
  const int tid = threadIdx.x, lane = tid & 31, w = tid >> 5;
  const int bh = blockIdx.z;                 // b*H + h
  const int b  = bh / HHEADS, h = bh % HHEADS;
  const int q0 = blockIdx.y * 128, kt0 = blockIdx.x * 128;
  const size_t hoff = (size_t)h * DKDIM;

#pragma unroll
  for (int i = tid; i < 128 * 64; i += 256) {
    const int r = i >> 6, c = i & 63;
    sA[i] = qh[(size_t)(b * LLEN + q0 + r) * (HHEADS * DKDIM) + hoff + c];
  }
#pragma unroll
  for (int i = tid; i < 128 * 64; i += 256) {
    const int n = i >> 6, c = i & 63;   // transpose into [d][n]
    sB[c * 128 + n] = kh[(size_t)(b * LLEN + kt0 + n) * (HHEADS * DKDIM) + hoff + c];
  }
  __syncthreads();

  const int wm = (w & 1) * 64, wn = (w >> 1) * 32;
  v8f acc[4][2];
#pragma unroll
  for (int mi = 0; mi < 4; ++mi)
#pragma unroll
    for (int ni = 0; ni < 2; ++ni) acc[mi][ni] = (v8f){0, 0, 0, 0, 0, 0, 0, 0};

#pragma unroll
  for (int k0 = 0; k0 < DKDIM; k0 += 32) {
    v16bf af[4], bfr[2];
#pragma unroll
    for (int mi = 0; mi < 4; ++mi) af[mi] = frag_a(sA + k0, wm + mi * 16, 64, lane);
#pragma unroll
    for (int ni = 0; ni < 2; ++ni) bfr[ni] = frag_b(sB + k0 * 128, wn + ni * 16, 128, lane);
#pragma unroll
    for (int mi = 0; mi < 4; ++mi)
#pragma unroll
      for (int ni = 0; ni < 2; ++ni) acc[mi][ni] = wmma_bf16(af[mi], bfr[ni], acc[mi][ni]);
  }

  const size_t base  = (size_t)bh * LLEN * LLEN;
  const int    cn    = kt0 + wn + (lane & 15);
  const int    rbase = q0 + wm + ((lane >> 4) << 3);
#pragma unroll
  for (int mi = 0; mi < 4; ++mi)
#pragma unroll
    for (int ni = 0; ni < 2; ++ni)
#pragma unroll
      for (int r = 0; r < 8; ++r)
        scores[base + (size_t)(rbase + mi * 16 + r) * LLEN + cn + ni * 16] =
            acc[mi][ni][r] * scale;
}

// ---------------- top-k threshold + masked softmax, one wave32 per row --------
__global__ __launch_bounds__(256) void topk_softmax_kernel(float* __restrict__ attn,
                                                           float* __restrict__ mask) {
  __shared__ float srow[8][LLEN];   // 64 KB
  const int tid = threadIdx.x, lane = tid & 31, w = tid >> 5;
  const size_t row  = (size_t)blockIdx.x * 8 + w;
  const size_t base = row * LLEN;
  float* sr = srow[w];

  for (int i = lane; i < LLEN; i += 32) sr[i] = attn[base + i];
  __syncthreads();

  float thr = __builtin_inff(), rowmax = -__builtin_inff();
  for (int it = 0; it < TOPKN; ++it) {          // 32nd-largest value = vk
    float loc = -__builtin_inff();
    for (int i = lane; i < LLEN; i += 32) {
      const float v = sr[i];
      if (v < thr) loc = fmaxf(loc, v);
    }
#pragma unroll
    for (int off = 16; off; off >>= 1) loc = fmaxf(loc, __shfl_xor(loc, off, 32));
    thr = loc;
    if (it == 0) rowmax = loc;
  }

  float s = 0.f;
  for (int i = lane; i < LLEN; i += 32) {
    const float v = sr[i];
    if (v >= thr) s += __expf(v - rowmax);
  }
#pragma unroll
  for (int off = 16; off; off >>= 1) s += __shfl_xor(s, off, 32);
  const float inv = 1.0f / s;

  for (int i = lane; i < LLEN; i += 32) {
    const float v    = sr[i];
    const bool  keep = v >= thr;
    attn[base + i] = keep ? __expf(v - rowmax) * inv : 0.f;
    mask[base + i] = keep ? 0.f : 1.f;
  }
}

// ---------------- ctx = attn @ Vh per (b,h): 128x64, K-loop over L ------------
__global__ __launch_bounds__(256) void attn_ctx_kernel(const float* __restrict__ attn,
                                                       const unsigned short* __restrict__ vh,
                                                       unsigned short* __restrict__ ctx) {
  __shared__ unsigned short sA[128 * 32];
  __shared__ unsigned short sB[32 * 64];
  const int tid = threadIdx.x, lane = tid & 31, w = tid >> 5;
  const int bh = blockIdx.y;
  const int b  = bh / HHEADS, h = bh % HHEADS;
  const int q0 = blockIdx.x * 128;
  const size_t arow = ((size_t)bh * LLEN + q0) * LLEN;

  v8f acc[4];
#pragma unroll
  for (int ni = 0; ni < 4; ++ni) acc[ni] = (v8f){0, 0, 0, 0, 0, 0, 0, 0};

  for (int k0 = 0; k0 < LLEN; k0 += 32) {
    __syncthreads();
#pragma unroll
    for (int i = tid; i < 128 * 32; i += 256) {
      const int r = i >> 5, c = i & 31;
      sA[i] = f32_to_bf16(attn[arow + (size_t)r * LLEN + k0 + c]);
    }
#pragma unroll
    for (int i = tid; i < 32 * 64; i += 256) {
      const int r = i >> 6, c = i & 63;
      sB[i] = vh[(size_t)(b * LLEN + k0 + r) * (HHEADS * DVDIM) + h * DVDIM + c];
    }
    __syncthreads();

    const v16bf a = frag_a(sA, w * 16, 32, lane);   // wave w owns rows w*16..w*16+15
#pragma unroll
    for (int ni = 0; ni < 4; ++ni) {
      const v16bf bfrag = frag_b(sB, ni * 16, 64, lane);
      acc[ni] = wmma_bf16(a, bfrag, acc[ni]);
    }
  }

  const int rbase = q0 + w * 16 + ((lane >> 4) << 3);
#pragma unroll
  for (int ni = 0; ni < 4; ++ni)
#pragma unroll
    for (int r = 0; r < 8; ++r)
      ctx[(size_t)(b * LLEN + rbase + r) * (HHEADS * DVDIM) + h * DVDIM + ni * 16 +
          (lane & 15)] = f32_to_bf16(acc[ni][r]);
}

// ---------------- residual + LayerNorm epilogue ------------------------------
__global__ __launch_bounds__(256) void ln_residual_kernel(
    const float* __restrict__ tmp, const float* __restrict__ resid,
    const float* __restrict__ gamma, const float* __restrict__ beta,
    float* __restrict__ out) {
  __shared__ float red[256];
  const int    tid  = threadIdx.x;
  const size_t base = (size_t)blockIdx.x * DDIM;

  float x[4], s = 0.f;
#pragma unroll
  for (int j = 0; j < 4; ++j) {
    const int c = tid + j * 256;
    x[j] = tmp[base + c] + resid[base + c];
    s += x[j];
  }
  red[tid] = s;
  __syncthreads();
  for (int st = 128; st; st >>= 1) {
    if (tid < st) red[tid] += red[tid + st];
    __syncthreads();
  }
  const float mu = red[0] * (1.0f / DDIM);
  __syncthreads();

  float sq = 0.f;
#pragma unroll
  for (int j = 0; j < 4; ++j) {
    const float d = x[j] - mu;
    sq += d * d;
  }
  red[tid] = sq;
  __syncthreads();
  for (int st = 128; st; st >>= 1) {
    if (tid < st) red[tid] += red[tid + st];
    __syncthreads();
  }
  const float rinv = rsqrtf(red[0] * (1.0f / DDIM) + 1e-6f);

#pragma unroll
  for (int j = 0; j < 4; ++j) {
    const int c = tid + j * 256;
    out[base + c] = (x[j] - mu) * rinv * gamma[c] + beta[c];
  }
}

// ---------------- host-side launcher ------------------------------------------
extern "C" void kernel_launch(void* const* d_in, const int* in_sizes, int n_in,
                              void* d_out, int out_size, void* d_ws, size_t ws_size,
                              hipStream_t stream) {
  (void)in_sizes; (void)n_in; (void)out_size; (void)ws_size;
  const float* q     = (const float*)d_in[0];
  const float* k     = (const float*)d_in[1];
  const float* v     = (const float*)d_in[2];
  const float* wq    = (const float*)d_in[3];
  const float* wk    = (const float*)d_in[4];
  const float* wv    = (const float*)d_in[5];
  const float* wo    = (const float*)d_in[6];
  const float* gamma = (const float*)d_in[7];
  const float* beta  = (const float*)d_in[8];

  const int M = BDIM * LLEN;        // 4096 tokens
  const int N = HHEADS * DKDIM;     // 1024

  float* out  = (float*)d_out;
  float* attn = out + (size_t)M * DDIM;
  float* mask = attn + (size_t)BDIM * HHEADS * LLEN * LLEN;

  unsigned short* qh  = (unsigned short*)d_ws;
  unsigned short* kh  = qh + (size_t)M * N;
  unsigned short* vh  = kh + (size_t)M * N;
  unsigned short* ctx = vh + (size_t)M * N;
  float*          tmp = (float*)(ctx + (size_t)M * N);

  const dim3 blk(256);
  const dim3 gproj(N / 128, M / 128);

  // Q/K/V projections (f32 in -> bf16 out, [b,l,h,dk] layout)
  gemm128x128<false, false><<<gproj, blk, 0, stream>>>(q, wq, qh, M, N, DDIM, DDIM, N, N);
  gemm128x128<false, false><<<gproj, blk, 0, stream>>>(k, wk, kh, M, N, DDIM, DDIM, N, N);
  gemm128x128<false, false><<<gproj, blk, 0, stream>>>(v, wv, vh, M, N, DDIM, DDIM, N, N);

  // scores = Qh Kh^T / sqrt(64), written into attn region of d_out
  const dim3 gsc(LLEN / 128, LLEN / 128, BDIM * HHEADS);
  attn_scores_kernel<<<gsc, blk, 0, stream>>>(qh, kh, attn, 0.125f);

  // top-k threshold + masked softmax in place; emits mask
  topk_softmax_kernel<<<dim3((BDIM * HHEADS * LLEN) / 8), blk, 0, stream>>>(attn, mask);

  // ctx = attn @ Vh  ([b,l,h,dv] bf16)
  attn_ctx_kernel<<<dim3(LLEN / 128, BDIM * HHEADS), blk, 0, stream>>>(attn, vh, ctx);

  // out-proj (bf16 A, f32 out) then residual + LayerNorm
  gemm128x128<true, true><<<gproj, blk, 0, stream>>>(ctx, wo, tmp, M, DDIM, N, N, DDIM, DDIM);
  ln_residual_kernel<<<dim3(M), blk, 0, stream>>>(tmp, q, gamma, beta, out);
}
